// ks_net_block3_16810501996768
// MI455X (gfx1250) — compile-verified, matrix operand order checked by reading, not run on hardware
//
#include <hip/hip_runtime.h>
#include <math.h>

typedef __attribute__((ext_vector_type(16))) _Float16 v16h;
typedef __attribute__((ext_vector_type(8)))  _Float16 v8h;
typedef __attribute__((ext_vector_type(8)))  float    v8f;

#define HW      128
#define LPIX    (HW * HW)      // 16384
#define DMODEL  16
#define DINNER  32
#define NSTATE  16
#define KDIR    4

// ---------------------------------------------------------------------------
// WMMA GEMM: C(M x N) = A(M x Kp) * B_packed [+ bias][+ act]
// A: f16 row-major, lda = Kp (multiple of 32, zero-padded), rows 16B-aligned.
// B: pre-packed into WMMA per-lane fragment layout by pack_wt:
//      [ntile][kchunk][lane 0..31][16 halves]  (Npad/16 tiles, Kp/32 chunks)
//    so each lane loads its whole fragment as one contiguous 32B v16h.
// One wave = one 16x16 tile. No EXEC manipulation before the WMMA.
// act: 0 none, 1 gelu(exact), 2 leaky-relu 0.01, 3 abs
// ---------------------------------------------------------------------------
__global__ void gemm_wmma_f16(const _Float16* __restrict__ A,
                              const _Float16* __restrict__ B,
                              float* __restrict__ C,
                              const float* __restrict__ bias,
                              int N, int Kp, int act,
                              long sA, long sB, long sC)
{
  const _Float16* Ab = A + (long)blockIdx.z * sA;
  const _Float16* Bb = B + (long)blockIdx.z * sB;
  float*          Cb = C + (long)blockIdx.z * sC;

  const int m0   = blockIdx.y * 16;
  const int lane = threadIdx.x & 31;
  const int half = lane >> 4;   // selects K sub-half per ISA 16-bit A/B layout
  const int idx  = lane & 15;   // A row / B column within tile
  const int nkc  = Kp >> 5;     // number of K chunks of 32

  v8f acc = {0.f, 0.f, 0.f, 0.f, 0.f, 0.f, 0.f, 0.f};
  const _Float16* arow = Ab + (long)(m0 + idx) * Kp + half * 8;
  const _Float16* bptr = Bb + ((long)blockIdx.x * nkc * 32 + lane) * 16;

  for (int kc = 0; kc < nkc; ++kc) {
    const int k0 = kc * 32;
    if (kc + 1 < nkc) __builtin_prefetch(arow + k0 + 32, 0, 0);
    // A fragment: two contiguous 16B runs -> v16h
    v8h alo = *(const v8h*)(arow + k0);
    v8h ahi = *(const v8h*)(arow + k0 + 16);
    v16h av = __builtin_shufflevector(alo, ahi, 0, 1, 2, 3, 4, 5, 6, 7,
                                                8, 9, 10, 11, 12, 13, 14, 15);
    // B fragment: one contiguous 32B load
    v16h bv = *(const v16h*)(bptr + (long)kc * 512);
    acc = __builtin_amdgcn_wmma_f32_16x16x32_f16(false, av, false, bv,
                                                 (short)0, acc, false, false);
  }

  const int ncol = blockIdx.x * 16 + idx;
  if (ncol >= N) return;                  // only pad columns drop out here
  float bval = bias ? bias[ncol] : 0.f;
#pragma unroll
  for (int v = 0; v < 8; ++v) {
    int m = m0 + v + half * 8;            // M is always a multiple of 16 here
    float r = acc[v] + bval;
    if      (act == 1) r = 0.5f * r * (1.f + erff(r * 0.70710678118f));
    else if (act == 2) r = (r > 0.f) ? r : 0.01f * r;
    else if (act == 3) r = fabsf(r);
    Cb[(long)m * N + ncol] = r;
  }
}

// Pack weight W(N,K) f32 row-major -> WMMA fragment layout f16, zero-padded
// to (Kp, Npad).  dst[i]: e=i&15, lane=(i>>4)&31, kchunk/ntile from i>>9.
__global__ void pack_wt(const float* __restrict__ W, _Float16* __restrict__ Bp,
                        int N, int K, int Kp, int Npad, long sW, long sB)
{
  long i = blockIdx.x * (long)blockDim.x + threadIdx.x;
  if (i >= (long)Kp * Npad) return;
  int e    = (int)(i & 15);
  int lane = (int)((i >> 4) & 31);
  long rest = i >> 9;
  int nkc  = Kp >> 5;
  int kc   = (int)(rest % nkc);
  int ntile = (int)(rest / nkc);
  int hi = e & 1, v = e >> 1;
  int kk = ((v < 4) ? 2 * v : 16 + 2 * (v - 4)) + ((lane >= 16) ? 8 : 0) + hi;
  int k = kc * 32 + kk;
  int n = ntile * 16 + (lane & 15);
  const float* Wb = W + (long)blockIdx.y * sW;
  _Float16*    Bb = Bp + (long)blockIdx.y * sB;
  Bb[i] = (_Float16)((k < K && n < N) ? Wb[(long)n * K + k] : 0.f);
}

// ---------------------------------------------------------------------------
// im2col for 3x3 SAME conv -> f16 (LPIX x Kp), column order (c,kh,kw),
// zero-padded to Kp. Strides select NCHW (sC=HW*HW,sH=HW,sW=1) or
// channels-last L x C (sC=1, sH=HW*C, sW=C).
// ---------------------------------------------------------------------------
__global__ void im2col3(const float* __restrict__ src, _Float16* __restrict__ dst,
                        int C, int sC, int sH, int sW, int Kp)
{
  int l = blockIdx.x * blockDim.x + threadIdx.x;
  if (l >= LPIX) return;
  int h = l >> 7, w = l & 127;
  _Float16* out = dst + (long)l * Kp;
  int k = 0;
  for (int c = 0; c < C; ++c) {
    const float* sc = src + (long)c * sC;
    for (int dh = -1; dh <= 1; ++dh) {
      int hh = h + dh;
      bool hv = (hh >= 0) && (hh < HW);
      for (int dw = -1; dw <= 1; ++dw) {
        int ww = w + dw;
        float v = (hv && ww >= 0 && ww < HW) ? sc[hh * sH + ww * sW] : 0.f;
        out[k++] = (_Float16)v;
      }
    }
  }
  for (; k < Kp; ++k) out[k] = (_Float16)0.f;
}

// LayerNorm over channels-last (L x C); optional f32 out, optional f16 out
// zero-padded to Kp columns (GEMM A operand).
__global__ void layernorm_lc(const float* __restrict__ x, const float* __restrict__ g,
                             const float* __restrict__ b, float* __restrict__ of32,
                             _Float16* __restrict__ of16, int C, int Kp)
{
  int l = blockIdx.x * blockDim.x + threadIdx.x;
  if (l >= LPIX) return;
  const float* xr = x + (long)l * C;
  float mu = 0.f;
  for (int c = 0; c < C; ++c) mu += xr[c];
  mu /= (float)C;
  float var = 0.f;
  for (int c = 0; c < C; ++c) { float d = xr[c] - mu; var += d * d; }
  float inv = rsqrtf(var / (float)C + 1e-5f);
  for (int c = 0; c < C; ++c) {
    float r = (xr[c] - mu) * inv * g[c] + b[c];
    if (of32) of32[(long)l * C + c] = r;
    if (of16) of16[(long)l * Kp + c] = (_Float16)r;
  }
  if (of16) for (int c = C; c < Kp; ++c) of16[(long)l * Kp + c] = (_Float16)0.f;
}

// Depthwise 3x3 SAME conv on xi = xz[:, 0:32] (channels-last L x 64) + SiLU
__global__ void dwconv_silu(const float* __restrict__ xz, const float* __restrict__ cw,
                            const float* __restrict__ cb, float* __restrict__ xc)
{
  long i = blockIdx.x * (long)blockDim.x + threadIdx.x;
  if (i >= (long)LPIX * DINNER) return;
  int d = (int)(i & 31);
  int l = (int)(i >> 5);
  int h = l >> 7, w = l & 127;
  float s = cb[d];
  for (int dh = -1; dh <= 1; ++dh) {
    int hh = h + dh; if (hh < 0 || hh >= HW) continue;
    for (int dw = -1; dw <= 1; ++dw) {
      int ww = w + dw; if (ww < 0 || ww >= HW) continue;
      s += xz[(long)(hh * HW + ww) * 64 + d] * cw[d * 9 + (dh + 1) * 3 + (dw + 1)];
    }
  }
  xc[i] = s / (1.f + __expf(-s));   // SiLU
}

// Build 4 scan orderings: (k, L, DINNER) f16.  k0=hw, k1=wh, k2/k3 reversed.
__global__ void build_xs(const float* __restrict__ xc, _Float16* __restrict__ xs)
{
  long i = blockIdx.x * (long)blockDim.x + threadIdx.x;
  if (i >= (long)LPIX * DINNER) return;
  int d = (int)(i & 31);
  int l = (int)(i >> 5);
  int h = l >> 7, w = l & 127;
  int lwh = w * HW + h;
  _Float16 v = (_Float16)xc[i];
  xs[((long)0 * LPIX + l)              * DINNER + d] = v;
  xs[((long)1 * LPIX + lwh)            * DINNER + d] = v;
  xs[((long)2 * LPIX + (LPIX - 1 - l)) * DINNER + d] = v;
  xs[((long)3 * LPIX + (LPIX - 1 - lwh)) * DINNER + d] = v;
}

// delta = softplus(dt_w[k,d] * x_dbl[k,l,0] + dt_b[k,d]); layout (k,L,32)
__global__ void delta_softplus(const float* __restrict__ x_dbl,
                               const float* __restrict__ dt_w,
                               const float* __restrict__ dt_b,
                               float* __restrict__ delta)
{
  long i = blockIdx.x * (long)blockDim.x + threadIdx.x;
  if (i >= (long)KDIR * LPIX * DINNER) return;
  int d  = (int)(i & 31);
  long kl = i >> 5;                  // k*L + l
  int k  = (int)(kl / LPIX);
  float dts = x_dbl[kl * 33 + 0];
  float v = dt_w[k * DINNER + d] * dts + dt_b[k * DINNER + d];
  delta[i] = (v > 20.f) ? v : log1pf(__expf(v));
}

// Selective scan: block = direction k (4), lane = channel d (32).
// 16 SSM states per lane in registers, sequential over L.
__global__ void sel_scan(const _Float16* __restrict__ xs, const float* __restrict__ delta,
                         const float* __restrict__ x_dbl, const float* __restrict__ A_log,
                         const float* __restrict__ Dskip, float* __restrict__ oy)
{
  int k = blockIdx.x;
  int d = threadIdx.x;
  float A[NSTATE], h[NSTATE];
#pragma unroll
  for (int n = 0; n < NSTATE; ++n) {
    A[n] = -__expf(A_log[(k * DINNER + d) * NSTATE + n]);
    h[n] = 0.f;
  }
  float Dk = Dskip[k * DINNER + d];
  long base = (long)k * LPIX;
  for (int l = 0; l < LPIX; ++l) {
    long kl = base + l;
    float u  = (float)xs[kl * DINNER + d];
    float dl = delta[kl * DINNER + d];
    const float* col = x_dbl + kl * 33;
    float du = dl * u;
    float acc = 0.f;
#pragma unroll
    for (int n = 0; n < NSTATE; ++n) {
      h[n] = __expf(dl * A[n]) * h[n] + du * col[1 + n];
      acc += h[n] * col[17 + n];
    }
    oy[kl * DINNER + d] = acc + Dk * u;
  }
}

// Combine 4 directions, LayerNorm(32) with out_norm, * SiLU(z) -> f16 (L x 32)
__global__ void combine_norm_silu(const float* __restrict__ oy, const float* __restrict__ xz,
                                  const float* __restrict__ g, const float* __restrict__ b,
                                  _Float16* __restrict__ out)
{
  int l = blockIdx.x * blockDim.x + threadIdx.x;
  if (l >= LPIX) return;
  int h = l >> 7, w = l & 127;
  int lwh = w * HW + h;
  float y[DINNER];
  float mu = 0.f;
  for (int d = 0; d < DINNER; ++d) {
    float v = oy[((long)0 * LPIX + l) * DINNER + d]
            + oy[((long)2 * LPIX + (LPIX - 1 - l)) * DINNER + d]
            + oy[((long)1 * LPIX + lwh) * DINNER + d]
            + oy[((long)3 * LPIX + (LPIX - 1 - lwh)) * DINNER + d];
    y[d] = v; mu += v;
  }
  mu /= (float)DINNER;
  float var = 0.f;
  for (int d = 0; d < DINNER; ++d) { float t = y[d] - mu; var += t * t; }
  float inv = rsqrtf(var / (float)DINNER + 1e-5f);
  for (int d = 0; d < DINNER; ++d) {
    float z  = xz[(long)l * 64 + 32 + d];
    float sz = z / (1.f + __expf(-z));
    out[(long)l * DINNER + d] = (_Float16)(((y[d] - mu) * inv * g[d] + b[d]) * sz);
  }
}

// out = a * s[c] + b                  (per (l,c))
__global__ void scale_add(const float* __restrict__ a, const float* __restrict__ s,
                          const float* __restrict__ b, float* __restrict__ out, int C)
{
  long i = blockIdx.x * (long)blockDim.x + threadIdx.x;
  if (i >= (long)LPIX * C) return;
  int c = (int)(i % C);
  out[i] = a[i] * s[c] + b[i];
}

// out = a * s[c] + b * att[c]
__global__ void scale_add2(const float* __restrict__ a, const float* __restrict__ s,
                           const float* __restrict__ b, const float* __restrict__ att,
                           float* __restrict__ out, int C)
{
  long i = blockIdx.x * (long)blockDim.x + threadIdx.x;
  if (i >= (long)LPIX * C) return;
  int c = (int)(i % C);
  out[i] = a[i] * s[c] + b[i] * att[c];
}

__global__ void add_ew(const float* __restrict__ a, const float* __restrict__ b,
                       float* __restrict__ o, long n)
{
  long i = blockIdx.x * (long)blockDim.x + threadIdx.x;
  if (i < n) o[i] = a[i] + b[i];
}

// Per-channel mean over L (one block per channel, LDS tree reduction)
__global__ void chan_mean(const float* __restrict__ x, float* __restrict__ mean, int C)
{
  __shared__ float red[256];
  int c = blockIdx.x;
  float s = 0.f;
  for (int l = threadIdx.x; l < LPIX; l += blockDim.x) s += x[(long)l * C + c];
  red[threadIdx.x] = s;
  __syncthreads();
  for (int off = 128; off > 0; off >>= 1) {
    if ((int)threadIdx.x < off) red[threadIdx.x] += red[threadIdx.x + off];
    __syncthreads();
  }
  if (threadIdx.x == 0) mean[c] = red[0] / (float)LPIX;
}

// att[c] = sigmoid(w2[c] * relu(sum_i w1[i]*mean[i] + b1) + b2[c])
__global__ void ca_gate(const float* __restrict__ mean, const float* __restrict__ w1,
                        const float* __restrict__ b1, const float* __restrict__ w2,
                        const float* __restrict__ b2, float* __restrict__ att)
{
  int c = threadIdx.x;
  float s = b1[0];
  for (int i = 0; i < DMODEL; ++i) s += w1[i] * mean[i];
  s = fmaxf(s, 0.f);
  float v = w2[c] * s + b2[c];
  att[c] = 1.f / (1.f + __expf(-v));
}

// channels-last (L x C) -> NCHW (C x L)
__global__ void lc_to_chw(const float* __restrict__ in, float* __restrict__ out, int C)
{
  long i = blockIdx.x * (long)blockDim.x + threadIdx.x;
  if (i >= (long)LPIX * C) return;
  int c = (int)(i % C);
  long l = i / C;
  out[(long)c * LPIX + l] = in[i];
}

// ---------------------------------------------------------------------------
extern "C" void kernel_launch(void* const* d_in, const int* in_sizes, int n_in,
                              void* d_out, int out_size, void* d_ws, size_t ws_size,
                              hipStream_t stream)
{
  (void)in_sizes; (void)n_in; (void)out_size; (void)ws_size;
  const long L = LPIX;
  char* ws = (char*)d_ws;
  size_t off = 0;
  auto arena = [&](size_t bytes) -> void* {
    off = (off + 255) & ~(size_t)255;
    void* p = ws + off;
    off += bytes;
    return p;
  };

  float*    fea   = (float*)arena(L * 16 * 4);
  float*    t     = (float*)arena(L * 16 * 4);
  float*    t0    = (float*)arena(L * 16 * 4);
  float*    xz    = (float*)arena(L * 64 * 4);
  float*    xc    = (float*)arena(L * 32 * 4);
  // BIG region: x_dbl | delta | oy (contiguous; aliased by im2col buffers)
  float*    x_dbl = (float*)arena((size_t)KDIR * L * 33 * 4);
  float*    delta = (float*)arena((size_t)KDIR * L * 32 * 4);
  float*    oy    = (float*)arena((size_t)KDIR * L * 32 * 4);
  _Float16* aln   = (_Float16*)arena(L * 32 * 2);
  _Float16* xs    = (_Float16*)arena((size_t)KDIR * L * 32 * 2);
  _Float16* ynorm = (_Float16*)arena(L * 32 * 2);
  float*    y2    = (float*)arena(L * 16 * 4);
  float*    xbuf  = (float*)arena(L * 16 * 4);
  float*    xln2  = (float*)arena(L * 16 * 4);
  float*    c1    = (float*)arena(L * 5 * 4);
  float*    c2    = (float*)arena(L * 16 * 4);
  float*    outLC = (float*)arena(L * 49 * 4);
  _Float16* bpack = (_Float16*)arena(65536 * 2);
  float*    mean  = (float*)arena(64 * 4);
  float*    att   = mean + 16;

  _Float16* col_small = (_Float16*)delta;  // Kp<=160  (5.2 MB <= 8 MB)
  _Float16* col_big   = (_Float16*)x_dbl;  // Kp=576  (18.9 MB <= 24.6 MB region)
  float*    u         = xz;                // L x 64, reused after blocks

  auto F = [&](int i) { return (const float*)d_in[i]; };

  auto pack = [&](const float* W, int N, int K, int Kp, int Npad,
                  int batch, long sW, long sB) {
    long n = (long)Kp * Npad;
    dim3 grid((unsigned)((n + 255) / 256), (unsigned)batch);
    pack_wt<<<grid, 256, 0, stream>>>(W, bpack, N, K, Kp, Npad, sW, sB);
  };
  auto gemm = [&](const _Float16* A, float* C, const float* bias,
                  int N, int Kp, int Npad, int act,
                  int batch, long sA, long sB, long sC) {
    dim3 grid((unsigned)(Npad / 16), (unsigned)(LPIX / 16), (unsigned)batch);
    gemm_wmma_f16<<<grid, 32, 0, stream>>>(A, bpack, C, bias, N, Kp, act,
                                           sA, sB, sC);
  };

  const int EW32 = (int)((L * 32) / 256);   // grids for elementwise kernels
  const int EW16 = (int)((L * 16) / 256);
  const int EWL  = (int)(L / 256);

  // --- one VSS block; param base index bi; input/output in `t` -------------
  auto vss = [&](int bi) {
    layernorm_lc<<<EWL, 256, 0, stream>>>(t, F(bi + 0), F(bi + 1), nullptr, aln, 16, 32);
    pack(F(bi + 2), 64, 16, 32, 64, 1, 0, 0);                       // in_proj (64,16)
    gemm(aln, xz, nullptr, 64, 32, 64, 0, 1, 0, 0, 0);
    dwconv_silu<<<EW32, 256, 0, stream>>>(xz, F(bi + 3), F(bi + 4), xc);
    build_xs<<<EW32, 256, 0, stream>>>(xc, xs);
    pack(F(bi + 5), 33, 32, 32, 48, 4, 33 * 32, 32 * 48);           // x_proj (4,33,32)
    gemm(xs, x_dbl, nullptr, 33, 32, 48, 0, 4, L * 32, 32 * 48, L * 33);
    delta_softplus<<<(int)((KDIR * L * 32) / 256), 256, 0, stream>>>(
        x_dbl, F(bi + 6), F(bi + 7), delta);
    sel_scan<<<KDIR, DINNER, 0, stream>>>(xs, delta, x_dbl, F(bi + 8), F(bi + 9), oy);
    combine_norm_silu<<<EWL, 256, 0, stream>>>(oy, xz, F(bi + 10), F(bi + 11), ynorm);
    pack(F(bi + 12), 16, 32, 32, 16, 1, 0, 0);                      // out_proj (16,32)
    gemm(ynorm, y2, nullptr, 16, 32, 16, 0, 1, 0, 0, 0);
    scale_add<<<EW16, 256, 0, stream>>>(t, F(bi + 13), y2, xbuf, 16);   // skip1
    layernorm_lc<<<EWL, 256, 0, stream>>>(xbuf, F(bi + 15), F(bi + 16), xln2, nullptr, 16, 0);
    // CAB: conv 16->5 (gelu), conv 5->16
    im2col3<<<EWL, 256, 0, stream>>>(xln2, col_small, 16, 1, HW * 16, 16, 160);
    pack(F(bi + 17), 5, 144, 160, 16, 1, 0, 0);
    gemm(col_small, c1, F(bi + 18), 5, 160, 16, 1, 1, 0, 0, 0);
    im2col3<<<EWL, 256, 0, stream>>>(c1, col_small, 5, 1, HW * 5, 5, 64);
    pack(F(bi + 19), 16, 45, 64, 16, 1, 0, 0);
    gemm(col_small, c2, F(bi + 20), 16, 64, 16, 0, 1, 0, 0, 0);
    // channel attention
    chan_mean<<<16, 256, 0, stream>>>(c2, mean, 16);
    ca_gate<<<1, 16, 0, stream>>>(mean, F(bi + 21), F(bi + 22), F(bi + 23), F(bi + 24), att);
    scale_add2<<<EW16, 256, 0, stream>>>(xbuf, F(bi + 14), c2, att, t, 16);  // skip2
  };

  auto conv16 = [&](const float* in, float* out, int wi, int biasi, int N, int act) {
    im2col3<<<EWL, 256, 0, stream>>>(in, col_small, 16, 1, HW * 16, 16, 160);
    pack(F(wi), N, 144, 160, N, 1, 0, 0);            // N is 16 or 64 (mult of 16)
    gemm(col_small, out, F(biasi), N, 160, N, act, 1, 0, 0, 0);
  };

  // ------------------- forward -------------------
  // conv_first: x NCHW (49,128,128) -> fea (L,16)
  im2col3<<<EWL, 256, 0, stream>>>((const float*)d_in[0], col_big, 49, LPIX, HW, 1, 448);
  pack(F(1), 16, 441, 448, 16, 1, 0, 0);
  gemm(col_big, fea, F(2), 16, 448, 16, 0, 1, 0, 0, 0);
  // pe_norm -> t
  layernorm_lc<<<EWL, 256, 0, stream>>>(fea, F(3), F(4), t, nullptr, 16, 0);

  // group 0: 1 block
  hipMemcpyAsync(t0, t, L * 16 * 4, hipMemcpyDeviceToDevice, stream);
  vss(5);
  conv16(t, y2, 30, 31, 16, 0);
  add_ew<<<EW16, 256, 0, stream>>>(y2, t0, t, L * 16);

  // group 1: 2 blocks
  hipMemcpyAsync(t0, t, L * 16 * 4, hipMemcpyDeviceToDevice, stream);
  vss(32);
  vss(57);
  conv16(t, y2, 82, 83, 16, 0);
  add_ew<<<EW16, 256, 0, stream>>>(y2, t0, t, L * 16);

  // tail: norm -> body conv + fea -> up conv (lrelu) -> last conv (abs)
  layernorm_lc<<<EWL, 256, 0, stream>>>(t, F(84), F(85), xln2, nullptr, 16, 0);
  conv16(xln2, y2, 86, 87, 16, 0);
  add_ew<<<EW16, 256, 0, stream>>>(y2, fea, xbuf, L * 16);
  conv16(xbuf, u, 88, 89, 64, 2);                       // up: 16->64, leaky relu
  im2col3<<<EWL, 256, 0, stream>>>(u, col_big, 64, 1, HW * 64, 64, 576);
  pack(F(90), 49, 576, 576, 64, 1, 0, 0);
  gemm(col_big, outLC, F(91), 49, 576, 64, 3, 1, 0, 0, 0);  // last: 64->49, abs
  lc_to_chw<<<(int)((L * 49 + 255) / 256), 256, 0, stream>>>(outLC, (float*)d_out, 49);
}
